// FFNN_encoder_86801289052512
// MI455X (gfx1250) — matrix operand order repaired and checked by reading, not compiled, not run on hardware
//
#include <hip/hip_runtime.h>

typedef __bf16 bf16;
typedef __attribute__((ext_vector_type(16))) __bf16 v16bf;
typedef __attribute__((ext_vector_type(8)))  __bf16 v8bf;
typedef __attribute__((ext_vector_type(8)))  float  v8f;

#define TOK 4096
#define DMODEL 1024
#define AF 2048
#define FF 4096
#define FIN 64
#define SEQ 2048
#define NHEAD 16
#define DH 128

static __device__ __forceinline__ v16bf cat8(v8bf lo, v8bf hi) {
  return __builtin_shufflevector(lo, hi, 0,1,2,3,4,5,6,7,8,9,10,11,12,13,14,15);
}
static __device__ __forceinline__ v8f wmma_bf16(v16bf a, v16bf b, v8f c) {
  return __builtin_amdgcn_wmma_f32_16x16x32_bf16(false, a, false, b, (short)0, c, false, false);
}
// CDNA5 async copy: 16B global -> LDS per lane, tracked by ASYNCcnt.
static __device__ __forceinline__ void async_copy_b128(const void* g, void* l) {
  unsigned loff = (unsigned)(uintptr_t)l;   // low 32 bits of flat shared addr = LDS offset
  asm volatile("global_load_async_to_lds_b128 %0, %1, off"
               :: "v"(loff), "v"(g) : "memory");
}
static __device__ __forceinline__ void async_wait0() {
  asm volatile("s_wait_asynccnt 0x0" ::: "memory");
}

// ---------------- conversion kernels ----------------
__global__ __launch_bounds__(256) void k_f32_to_bf16(const float* __restrict__ in,
                                                     bf16* __restrict__ out, int n) {
  int i = blockIdx.x * 256 + threadIdx.x;
  if (i < n) out[i] = (bf16)in[i];
}

// in[K][N] fp32 -> out[N][K] bf16
__global__ __launch_bounds__(256) void k_transpose_bf16(const float* __restrict__ in,
                                                        bf16* __restrict__ out, int K, int N) {
  __shared__ float tile[32][33];
  int k0 = blockIdx.y * 32, n0 = blockIdx.x * 32;
  int tx = threadIdx.x & 31, ty = threadIdx.x >> 5;
  for (int r = ty; r < 32; r += 8) {
    int k = k0 + r, n = n0 + tx;
    tile[r][tx] = (k < K && n < N) ? in[(size_t)k * N + n] : 0.f;
  }
  __syncthreads();
  for (int r = ty; r < 32; r += 8) {
    int n = n0 + r, k = k0 + tx;
    if (n < N && k < K) out[(size_t)n * K + k] = (bf16)tile[tx][r];
  }
}

// ---------------- WMMA GEMM: C = A[M][K] * Bt[N][K]^T + bias ----------------
#define BM 128
#define BN 64
#define BK 64
#define LDT 72   // BK + 8 pad (bf16 units); row stride 144B = 9*16B

template<bool RELU, bool WF, bool WB>
__global__ __launch_bounds__(256) void k_gemm_bf16(
    const bf16* __restrict__ A, const bf16* __restrict__ Bt,
    const float* __restrict__ bias,
    float* __restrict__ outF, bf16* __restrict__ outB,
    int M, int N, int K) {
  __shared__ __align__(16) bf16 As[BM * LDT];
  __shared__ __align__(16) bf16 Bs[BN * LDT];
  const int tid = threadIdx.x;
  const int wave = tid >> 5, lane = tid & 31;
  const int wm = wave >> 1, wn = wave & 1;
  const int kh = lane >> 4, ln = lane & 15;
  const int m0 = blockIdx.y * BM, n0 = blockIdx.x * BN;

  v8f zf = {0.f,0.f,0.f,0.f,0.f,0.f,0.f,0.f};
  v8f acc[2][2];
  for (int i = 0; i < 2; i++) for (int j = 0; j < 2; j++) acc[i][j] = zf;

  for (int kt = 0; kt < K; kt += BK) {
    // A tile: 128x64 bf16 = 1024 x 16B, 4 per thread (async -> LDS)
#pragma unroll
    for (int r = 0; r < 4; r++) {
      int idx = tid + 256 * r;
      int row = idx >> 3, c8 = idx & 7;
      async_copy_b128(A + (size_t)(m0 + row) * K + kt + c8 * 8, &As[row * LDT + c8 * 8]);
    }
    // B tile: 64x64 = 512 x 16B, 2 per thread
#pragma unroll
    for (int r = 0; r < 2; r++) {
      int idx = tid + 256 * r;
      int row = idx >> 3, c8 = idx & 7;
      async_copy_b128(Bt + (size_t)(n0 + row) * K + kt + c8 * 8, &Bs[row * LDT + c8 * 8]);
    }
    async_wait0();
    __syncthreads();

#pragma unroll
    for (int ks = 0; ks < 2; ks++) {
      v16bf afr[2], bfr[2];
#pragma unroll
      for (int mi = 0; mi < 2; mi++) {
        const bf16* ap = &As[(wm * 32 + mi * 16 + ln) * LDT + ks * 32];
        afr[mi] = cat8(*(const v8bf*)(ap + kh * 16), *(const v8bf*)(ap + kh * 16 + 8));
      }
#pragma unroll
      for (int ni = 0; ni < 2; ni++) {
        const bf16* bp = &Bs[(wn * 32 + ni * 16 + ln) * LDT + ks * 32 + kh * 16];
        bfr[ni] = cat8(*(const v8bf*)bp, *(const v8bf*)(bp + 8));
      }
#pragma unroll
      for (int mi = 0; mi < 2; mi++)
#pragma unroll
        for (int ni = 0; ni < 2; ni++)
          acc[mi][ni] = wmma_bf16(afr[mi], bfr[ni], acc[mi][ni]);
    }
    __syncthreads();
  }

#pragma unroll
  for (int mi = 0; mi < 2; mi++)
#pragma unroll
    for (int ni = 0; ni < 2; ni++) {
      int gn = n0 + wn * 32 + ni * 16 + ln;
      float bv = bias[gn];
#pragma unroll
      for (int v = 0; v < 8; v++) {
        int gm = m0 + wm * 32 + mi * 16 + kh * 8 + v;
        float val = acc[mi][ni][v] + bv;
        if (RELU) val = fmaxf(val, 0.f);
        size_t o = (size_t)gm * N + gn;
        if (WF) outF[o] = val;
        if (WB) outB[o] = (bf16)val;
      }
    }
}

// ---------------- LayerNorm (+timing signal or +residual) ----------------
__global__ __launch_bounds__(256) void k_ln(
    const float* __restrict__ xin, const float* __restrict__ resid,
    const float* __restrict__ gamma, const float* __restrict__ beta,
    float* __restrict__ outF, bf16* __restrict__ outB, int timing) {
  int t = blockIdx.x;
  int s = t & (SEQ - 1);
  const float* xrow = xin + (size_t)t * DMODEL;
  const float* rrow = resid ? resid + (size_t)t * DMODEL : nullptr;
  float vals[4], sum = 0.f, sq = 0.f;
#pragma unroll
  for (int i = 0; i < 4; i++) {
    int c = threadIdx.x + i * 256;
    float x = xrow[c];
    if (timing) {
      int idx = (c < 512) ? c : c - 512;
      float inv = __expf(-(float)idx * 0.01802414946f);   // ln(10000)/511
      float ang = (float)s * inv;
      x += (c < 512) ? __sinf(ang) : __cosf(ang);
    } else if (rrow) {
      x += rrow[c];
    }
    vals[i] = x; sum += x; sq += x * x;
  }
#pragma unroll
  for (int off = 16; off; off >>= 1) {
    sum += __shfl_xor(sum, off, 32);
    sq  += __shfl_xor(sq,  off, 32);
  }
  __shared__ float rs[8], rq[8], mu_s, rstd_s;
  if ((threadIdx.x & 31) == 0) { rs[threadIdx.x >> 5] = sum; rq[threadIdx.x >> 5] = sq; }
  __syncthreads();
  if (threadIdx.x == 0) {
    float S = 0.f, Q = 0.f;
    for (int i = 0; i < 8; i++) { S += rs[i]; Q += rq[i]; }
    float mu = S / (float)DMODEL;
    float var = Q / (float)DMODEL - mu * mu;
    mu_s = mu; rstd_s = rsqrtf(var + 1e-3f);
  }
  __syncthreads();
  float mu = mu_s, rstd = rstd_s;
#pragma unroll
  for (int i = 0; i < 4; i++) {
    int c = threadIdx.x + i * 256;
    float y = (vals[i] - mu) * rstd * gamma[c] + beta[c];
    size_t o = (size_t)t * DMODEL + c;
    outF[o] = y;
    if (outB) outB[o] = (bf16)y;
  }
}

// ---------------- Flash attention (causal, WMMA) ----------------
#define BQ 64
#define BKV 64
#define QLD 136   // DH + 8 pad; 272B rows
#define PLD 72
#define VLD 72

__global__ __launch_bounds__(256) void k_attn(
    const bf16* __restrict__ Qg, const bf16* __restrict__ Kg,
    const bf16* __restrict__ Vg, bf16* __restrict__ Og) {
  __shared__ __align__(16) bf16 Qs[BQ * QLD];
  __shared__ __align__(16) bf16 Ks[BKV * QLD];
  __shared__ __align__(16) bf16 Vts[DH * VLD];
  __shared__ __align__(16) bf16 Ps[BQ * PLD];
  __shared__ float mState[BQ], lState[BQ], fac[BQ];
  __shared__ float partMax[2][BQ], partSum[2][BQ];

  const int tid = threadIdx.x, wave = tid >> 5, lane = tid & 31;
  const int wm = wave >> 1, wn = wave & 1;
  const int kh = lane >> 4, ln = lane & 15;
  const int qt = blockIdx.x, h = blockIdx.y, b = blockIdx.z;
  const int q0 = qt * BQ;
  const size_t baseRow = (size_t)b * SEQ;
  const int colBase = h * DH;
  const float scale = 0.022097086912079608f;  // 1/sqrt(2048)

  // Q tile 64x128 via async copy
#pragma unroll
  for (int r = 0; r < 4; r++) {
    int idx = tid + 256 * r;
    int row = idx >> 4, c8 = idx & 15;
    async_copy_b128(Qg + (baseRow + q0 + row) * AF + colBase + c8 * 8,
                    &Qs[row * QLD + c8 * 8]);
  }
  if (tid < BQ) { mState[tid] = -3.0e38f; lState[tid] = 0.f; }
  async_wait0();
  __syncthreads();

  v8f zf = {0.f,0.f,0.f,0.f,0.f,0.f,0.f,0.f};
  v8f Oacc[4];
#pragma unroll
  for (int i = 0; i < 4; i++) Oacc[i] = zf;

  for (int kb = 0; kb <= qt; kb++) {
    int k0 = kb * BKV;
#pragma unroll
    for (int r = 0; r < 4; r++) {
      int idx = tid + 256 * r;
      int row = idx >> 4, c8 = idx & 15;
      async_copy_b128(Kg + (baseRow + k0 + row) * AF + colBase + c8 * 8,
                      &Ks[row * QLD + c8 * 8]);
      union { uint4 u; bf16 e[8]; } vv;
      vv.u = *(const uint4*)(Vg + (baseRow + k0 + row) * AF + colBase + c8 * 8);
#pragma unroll
      for (int e = 0; e < 8; e++) Vts[(c8 * 8 + e) * VLD + row] = vv.e[e];
    }
    async_wait0();
    __syncthreads();

    // S = Q K^T : M=64, N=64, Kdim=128
    v8f sacc[2] = { zf, zf };
#pragma unroll
    for (int ks = 0; ks < 4; ks++) {
      const bf16* ap = &Qs[(wm * 16 + ln) * QLD + ks * 32];
      v16bf afr = cat8(*(const v8bf*)(ap + kh * 16), *(const v8bf*)(ap + kh * 16 + 8));
#pragma unroll
      for (int ni = 0; ni < 2; ni++) {
        const bf16* bp = &Ks[(wn * 32 + ni * 16 + ln) * QLD + ks * 32 + kh * 16];
        v16bf bfr = cat8(*(const v8bf*)bp, *(const v8bf*)(bp + 8));
        sacc[ni] = wmma_bf16(afr, bfr, sacc[ni]);
      }
    }

    // scale + causal mask (diagonal block only) + row max
    bool diag = (kb == qt);
    float lmax[8];
#pragma unroll
    for (int v = 0; v < 8; v++) {
      int row = wm * 16 + kh * 8 + v;
#pragma unroll
      for (int ni = 0; ni < 2; ni++) {
        float s = sacc[ni][v] * scale;
        if (diag) {
          int key = wn * 32 + ni * 16 + ln;
          if (key > row) s = -3.0e38f;
        }
        sacc[ni][v] = s;
      }
      lmax[v] = fmaxf(sacc[0][v], sacc[1][v]);
    }
#pragma unroll
    for (int off = 1; off < 16; off <<= 1)
#pragma unroll
      for (int v = 0; v < 8; v++) lmax[v] = fmaxf(lmax[v], __shfl_xor(lmax[v], off, 32));
    if (ln == 0)
#pragma unroll
      for (int v = 0; v < 8; v++) partMax[wn][wm * 16 + kh * 8 + v] = lmax[v];
    __syncthreads();

    if (tid < BQ) {
      float mOld = mState[tid];
      float mNew = fmaxf(mOld, fmaxf(partMax[0][tid], partMax[1][tid]));
      mState[tid] = mNew;
      fac[tid] = __expf(mOld - mNew);
    }
    __syncthreads();

    // P = exp(S - m), rescale O, row sums
    float lsum[8];
#pragma unroll
    for (int v = 0; v < 8; v++) {
      int row = wm * 16 + kh * 8 + v;
      float m = mState[row], f = fac[row];
      float p0 = __expf(sacc[0][v] - m);
      float p1 = __expf(sacc[1][v] - m);
      Ps[row * PLD + wn * 32 + ln] = (bf16)p0;
      Ps[row * PLD + wn * 32 + 16 + ln] = (bf16)p1;
      lsum[v] = p0 + p1;
#pragma unroll
      for (int nt = 0; nt < 4; nt++) Oacc[nt][v] *= f;
    }
#pragma unroll
    for (int off = 1; off < 16; off <<= 1)
#pragma unroll
      for (int v = 0; v < 8; v++) lsum[v] += __shfl_xor(lsum[v], off, 32);
    if (ln == 0)
#pragma unroll
      for (int v = 0; v < 8; v++) partSum[wn][wm * 16 + kh * 8 + v] = lsum[v];
    __syncthreads();
    if (tid < BQ) lState[tid] = lState[tid] * fac[tid] + partSum[0][tid] + partSum[1][tid];

    // O += P V : M=64, N=128, Kdim=64
#pragma unroll
    for (int ks = 0; ks < 2; ks++) {
      const bf16* ap = &Ps[(wm * 16 + ln) * PLD + ks * 32];
      v16bf afr = cat8(*(const v8bf*)(ap + kh * 16), *(const v8bf*)(ap + kh * 16 + 8));
#pragma unroll
      for (int nt = 0; nt < 4; nt++) {
        const bf16* bp = &Vts[(wn * 64 + nt * 16 + ln) * VLD + ks * 32 + kh * 16];
        v16bf bfr = cat8(*(const v8bf*)bp, *(const v8bf*)(bp + 8));
        Oacc[nt] = wmma_bf16(afr, bfr, Oacc[nt]);
      }
    }
    __syncthreads();
  }

#pragma unroll
  for (int v = 0; v < 8; v++) {
    int row = wm * 16 + kh * 8 + v;
    float invl = 1.f / lState[row];
    size_t orow = (baseRow + q0 + row) * AF + colBase;
#pragma unroll
    for (int nt = 0; nt < 4; nt++) {
      int dcol = wn * 64 + nt * 16 + ln;
      Og[orow + dcol] = (bf16)(Oacc[nt][v] * invl);
    }
  }
}

// ---------------- host side ----------------
extern "C" void kernel_launch(void* const* d_in, const int* in_sizes, int n_in,
                              void* d_out, int out_size, void* d_ws, size_t ws_size,
                              hipStream_t stream) {
  const float* x     = (const float*)d_in[0];
  const float* Wemb  = (const float*)d_in[1];
  const float* bemb  = (const float*)d_in[2];
  const float* g0    = (const float*)d_in[3];
  const float* beta0 = (const float*)d_in[4];
  const float* Wq = (const float*)d_in[5];   const float* bq = (const float*)d_in[6];
  const float* Wk = (const float*)d_in[7];   const float* bk = (const float*)d_in[8];
  const float* Wv = (const float*)d_in[9];   const float* bv = (const float*)d_in[10];
  const float* Wo = (const float*)d_in[11];  const float* bo = (const float*)d_in[12];
  const float* g1 = (const float*)d_in[13];  const float* beta1 = (const float*)d_in[14];
  const float* W1 = (const float*)d_in[15];  const float* b1 = (const float*)d_in[16];
  const float* W2 = (const float*)d_in[17];  const float* b2 = (const float*)d_in[18];
  const float* g2 = (const float*)d_in[19];  const float* beta2 = (const float*)d_in[20];
  float* out = (float*)d_out;

  char* w = (char*)d_ws;
  size_t off = 0;
  auto take = [&](size_t bytes) -> char* {
    char* p = w + off; off = (off + bytes + 255) & ~(size_t)255; return p;
  };
  bf16* xb    = (bf16*)take((size_t)TOK * FIN * 2);
  bf16* WembT = (bf16*)take((size_t)DMODEL * FIN * 2);
  bf16* WqT   = (bf16*)take((size_t)AF * DMODEL * 2);
  bf16* WkT   = (bf16*)take((size_t)AF * DMODEL * 2);
  bf16* WvT   = (bf16*)take((size_t)AF * DMODEL * 2);
  bf16* WoT   = (bf16*)take((size_t)DMODEL * AF * 2);
  bf16* W1T   = (bf16*)take((size_t)FF * DMODEL * 2);
  bf16* W2T   = (bf16*)take((size_t)DMODEL * FF * 2);
  float* h0   = (float*)take((size_t)TOK * DMODEL * 4);
  bf16*  h0b  = (bf16*)take((size_t)TOK * DMODEL * 2);
  float* h1   = (float*)take((size_t)TOK * DMODEL * 4);
  bf16*  h1b  = (bf16*)take((size_t)TOK * DMODEL * 2);
  float* tmp  = (float*)take((size_t)TOK * DMODEL * 4);
  bf16*  qb   = (bf16*)take((size_t)TOK * AF * 2);   // qb..kb region reused as ffn1
  bf16*  kb   = (bf16*)take((size_t)TOK * AF * 2);
  bf16*  vb   = (bf16*)take((size_t)TOK * AF * 2);
  bf16*  ctxb = (bf16*)take((size_t)TOK * AF * 2);
  bf16*  ffn1b = qb;  // 4096x4096 bf16 fits in q+k region (dead after attention)

  // --- conversions ---
  k_f32_to_bf16<<<(TOK * FIN + 255) / 256, 256, 0, stream>>>(x, xb, TOK * FIN);
  k_transpose_bf16<<<dim3(DMODEL / 32, FIN / 32), 256, 0, stream>>>(Wemb, WembT, FIN, DMODEL);
  k_transpose_bf16<<<dim3(AF / 32, DMODEL / 32), 256, 0, stream>>>(Wq, WqT, DMODEL, AF);
  k_transpose_bf16<<<dim3(AF / 32, DMODEL / 32), 256, 0, stream>>>(Wk, WkT, DMODEL, AF);
  k_transpose_bf16<<<dim3(AF / 32, DMODEL / 32), 256, 0, stream>>>(Wv, WvT, DMODEL, AF);
  k_transpose_bf16<<<dim3(DMODEL / 32, AF / 32), 256, 0, stream>>>(Wo, WoT, AF, DMODEL);
  k_transpose_bf16<<<dim3(FF / 32, DMODEL / 32), 256, 0, stream>>>(W1, W1T, DMODEL, FF);
  k_transpose_bf16<<<dim3(DMODEL / 32, FF / 32), 256, 0, stream>>>(W2, W2T, FF, DMODEL);

  // --- embedding: relu(x Wemb + b) -> tmp; +timing, LN -> h0/h0b ---
  k_gemm_bf16<true, true, false><<<dim3(DMODEL / BN, TOK / BM), 256, 0, stream>>>(
      xb, WembT, bemb, tmp, nullptr, TOK, DMODEL, FIN);
  k_ln<<<TOK, 256, 0, stream>>>(tmp, nullptr, g0, beta0, h0, h0b, 1);

  // --- QKV projections (bf16 outputs) ---
  k_gemm_bf16<false, false, true><<<dim3(AF / BN, TOK / BM), 256, 0, stream>>>(
      h0b, WqT, bq, nullptr, qb, TOK, AF, DMODEL);
  k_gemm_bf16<false, false, true><<<dim3(AF / BN, TOK / BM), 256, 0, stream>>>(
      h0b, WkT, bk, nullptr, kb, TOK, AF, DMODEL);
  k_gemm_bf16<false, false, true><<<dim3(AF / BN, TOK / BM), 256, 0, stream>>>(
      h0b, WvT, bv, nullptr, vb, TOK, AF, DMODEL);

  // --- causal flash attention ---
  k_attn<<<dim3(SEQ / BQ, NHEAD, 2), 256, 0, stream>>>(qb, kb, vb, ctxb);

  // --- output projection + residual + LN ---
  k_gemm_bf16<false, true, false><<<dim3(DMODEL / BN, TOK / BM), 256, 0, stream>>>(
      ctxb, WoT, bo, tmp, nullptr, TOK, DMODEL, AF);
  k_ln<<<TOK, 256, 0, stream>>>(tmp, h0, g1, beta1, h1, h1b, 0);

  // --- FFN ---
  k_gemm_bf16<true, false, true><<<dim3(FF / BN, TOK / BM), 256, 0, stream>>>(
      h1b, W1T, b1, nullptr, ffn1b, TOK, FF, DMODEL);
  k_gemm_bf16<false, true, false><<<dim3(DMODEL / BN, TOK / BM), 256, 0, stream>>>(
      ffn1b, W2T, b2, tmp, nullptr, TOK, DMODEL, FF);
  k_ln<<<TOK, 256, 0, stream>>>(tmp, h1, g2, beta2, out, nullptr, 0);

  (void)in_sizes; (void)n_in; (void)out_size; (void)ws_size;
}
